// MultiAttention_24790551233249
// MI455X (gfx1250) — matrix operand order repaired and checked by reading, not compiled, not run on hardware
//
#include <hip/hip_runtime.h>
#include <hip/hip_bf16.h>

// MultiHeadAttention for MI455X (gfx1250, wave32, WMMA f16 16x16x32, f32 acc).
// Prep:  x->f16; Wq/Wk/Wv/Wc -> f16 transposed (N-major).
// Main:  all GEMM kernels double-buffer their shared B-operand slabs through
//        LDS with CDNA5 async copies (global_load_async_to_lds_b128 +
//        s_wait_asynccnt). Ping-pong is a loop-carried pointer swap with
//        unrolling disabled so accumulators stay pinned in one register range
//        (no cross-body v_dual_mov shuffles).

#define SEQ 4096
#define DIM 1024
#define NH  16
#define DKH 64
#define DVH 64

typedef __attribute__((ext_vector_type(16))) _Float16 v16h;
typedef __attribute__((ext_vector_type(8)))  _Float16 v8h;
typedef __attribute__((ext_vector_type(8)))  float    v8f;
typedef _Float16 h16;

#define WMMA_F16(A, B, C) \
  __builtin_amdgcn_wmma_f32_16x16x32_f16(false, (A), false, (B), (short)0, (C), false, false)

// ---------------- CDNA5 async copy: global -> LDS (ASYNCcnt) ----------------
// Per-lane 16-byte copy; LDS offset = low 32 bits of the flat shared pointer.
__device__ __forceinline__ void async_copy_b128(const h16* lds_dst, const h16* gsrc) {
  unsigned loff = (unsigned)(size_t)lds_dst;
  asm volatile("global_load_async_to_lds_b128 %0, %1, off"
               :: "v"(loff), "v"(gsrc) : "memory");
}
__device__ __forceinline__ void wait_async0() {
  asm volatile("s_wait_asynccnt 0" ::: "memory");
}

// Stage a contiguous 4KB slab (256 x 16B chunks) with 128 threads.
__device__ __forceinline__ void stage_linear(h16* lds, const h16* g, int tid) {
#pragma unroll
  for (int i = 0; i < 2; ++i) {
    int c = tid + i * 128;
    async_copy_b128(lds + c * 8, g + c * 8);
  }
}
// Stage 64 rows x 32 halves (64B rows, global row stride g_ld halves) -> 4KB.
__device__ __forceinline__ void stage_rows(h16* lds, const h16* g, size_t g_ld,
                                           int tid) {
#pragma unroll
  for (int i = 0; i < 2; ++i) {
    int c = tid + i * 128;              // chunk id; 4 chunks per 64B row
    int row = c >> 2, sub = c & 3;
    async_copy_b128(lds + c * 8, g + (size_t)row * g_ld + sub * 8);
  }
}

// ---------------- WMMA operand loaders (wave32) ----------------
// A layout 16x32 (MxK): lane<16 row=lane, halves 0..15 = K kb+0..15 with
// kb=0; lanes 16..31 same rows with kb=16.
__device__ __forceinline__ v16h load_a_f16(const h16* __restrict__ base, int ld,
                                           int row0, int k0, int lane) {
  int m  = lane & 15;
  int kb = (lane < 16) ? 0 : 16;
  const h16* p = base + (size_t)(row0 + m) * ld + k0 + kb;
  v8h lo = *(const v8h*)p;
  v8h hi = *(const v8h*)(p + 8);
  return __builtin_shufflevector(lo, hi, 0, 1, 2, 3, 4, 5, 6, 7,
                                 8, 9, 10, 11, 12, 13, 14, 15);
}
// B layout 32x16 (KxN), N-major source src[n][k]: contiguous 32B per lane.
__device__ __forceinline__ v16h load_b_nmajor_f16(const h16* base, int ld,
                                                  int k0, int n0, int lane) {
  int n  = lane & 15;
  int kb = (lane < 16) ? 0 : 16;
  return *(const v16h*)(base + (size_t)(n0 + n) * ld + k0 + kb);
}

// ---------------- Prep kernels ----------------
__global__ void cvt_x_kernel(const float* __restrict__ src, h16* __restrict__ dst) {
  size_t i = (size_t)blockIdx.x * 256 + threadIdx.x;
  dst[i] = (h16)src[i];
}
// dst[c][r] (f16) = src[r][c] (f32); blockIdx.z selects a head slab.
__global__ void tr_cvt_kernel(const float* __restrict__ src, h16* __restrict__ dst,
                              int R, int C, size_t srcStride, size_t dstStride) {
  __shared__ h16 tile[32][33];
  const float* s = src + (size_t)blockIdx.z * srcStride;
  h16* d = dst + (size_t)blockIdx.z * dstStride;
  int c0 = blockIdx.x * 32, r0 = blockIdx.y * 32;
  int tx = threadIdx.x & 31, ty = threadIdx.x >> 5;
#pragma unroll
  for (int i = ty; i < 32; i += 8)
    tile[i][tx] = (h16)s[(size_t)(r0 + i) * C + c0 + tx];
  __syncthreads();
#pragma unroll
  for (int i = ty; i < 32; i += 8)
    d[(size_t)(c0 + i) * R + r0 + tx] = tile[tx][i];
}

// ---------------- Kernel 1: QKV projection (one matrix per grid.z) --------
// block = 4 waves (one 16-row S tile each), grid = (S/64, H, 3).
// Weight slab [64 dk x 32 d] double-buffered in LDS via async copies.
__global__ void qkv_proj_kernel(const h16* __restrict__ x16,
                                const h16* __restrict__ WqT,
                                const h16* __restrict__ WkT,
                                const h16* __restrict__ WvT,
                                h16* __restrict__ qh, h16* __restrict__ kh,
                                h16* __restrict__ vT) {
  __shared__ h16 wbuf[2][64 * 32];  // 8 KB ping-pong
  const int tid  = threadIdx.x;
  const int lane = tid & 31;
  const int wave = tid >> 5;
  const int s0 = blockIdx.x * 64 + wave * 16;
  const int h  = blockIdx.y;
  const int mat = blockIdx.z;  // 0=q, 1=k, 2=v
  const h16* wsel = (mat == 0) ? WqT : (mat == 1) ? WkT : WvT;
  const h16* wt = wsel + (size_t)h * DKH * DIM;  // [dk][d]  (N-major)

  v8f acc[4] = {};
  h16* cur = wbuf[0];
  h16* nxt = wbuf[1];
  stage_rows(cur, wt, DIM, tid);
#pragma clang loop unroll(disable)
  for (int kk = 0; kk < DIM; kk += 32) {
    wait_async0();
    __syncthreads();
    if (kk + 32 < DIM) stage_rows(nxt, wt + kk + 32, DIM, tid);

    v16h a = load_a_f16(x16, DIM, s0, kk, lane);
    v16h b[4];
#pragma unroll
    for (int t = 0; t < 4; ++t)
      b[t] = load_b_nmajor_f16(cur, 32, 0, t * 16, lane);
#pragma unroll
    for (int t = 0; t < 4; ++t)
      acc[t] = WMMA_F16(a, b[t], acc[t]);
    __syncthreads();  // all waves done reading before buffer is restaged
    h16* tmp = cur; cur = nxt; nxt = tmp;
  }
  // C layout: vgpr r -> row moff+r, col = lane&15.
  const int n = lane & 15;
  const int moff = (lane < 16) ? 0 : 8;
  if (mat == 2) {
    // v stored transposed: vT[h][dv][s] -> contiguous B reads in attention.
#pragma unroll
    for (int t = 0; t < 4; ++t)
#pragma unroll
      for (int r = 0; r < 8; ++r)
        vT[(size_t)(h * DVH + t * 16 + n) * SEQ + s0 + moff + r] = (h16)acc[t][r];
  } else {
    h16* dst = (mat == 0) ? qh : kh;
#pragma unroll
    for (int t = 0; t < 4; ++t)
#pragma unroll
      for (int r = 0; r < 8; ++r)
        dst[((size_t)h * SEQ + s0 + moff + r) * DKH + t * 16 + n] = (h16)acc[t][r];
  }
}

// ---------------- Kernel 2: flash attention ----------------
// block = 4 waves (one 16-query tile each), grid = (S/64, H).
// K slab [32 keys x 64 dk] and V slab [64 dv x 32 keys] double-buffered in LDS.
__global__ void attn_kernel(const h16* __restrict__ qh, const h16* __restrict__ kh,
                            const h16* __restrict__ vT, h16* __restrict__ cat) {
  __shared__ h16 kbuf[2][32 * 64];   // 8 KB
  __shared__ h16 vbuf[2][64 * 32];   // 8 KB
  __shared__ h16 pbuf[4][16 * 32];   // per-wave C->A transpose bounce, 4 KB
  const int tid  = threadIdx.x;
  const int lane = tid & 31;
  const int wave = tid >> 5;
  const int s0 = blockIdx.x * 64 + wave * 16;
  const int h  = blockIdx.y;
  const h16* qb = qh + (size_t)h * SEQ * DKH;
  const h16* kb = kh + (size_t)h * SEQ * DKH;
  const h16* vb = vT + (size_t)h * DVH * SEQ;
  h16* pw = pbuf[wave];

  const v16h aq0 = load_a_f16(qb, DKH, s0, 0,  lane);
  const v16h aq1 = load_a_f16(qb, DKH, s0, 32, lane);

  float mrow[8], lrow[8];
  v8f o[4] = {};
#pragma unroll
  for (int r = 0; r < 8; ++r) { mrow[r] = -3.0e38f; lrow[r] = 0.0f; }

  const int n = lane & 15;
  const int moff = (lane < 16) ? 0 : 8;

  h16* kcur = kbuf[0]; h16* knxt = kbuf[1];
  h16* vcur = vbuf[0]; h16* vnxt = vbuf[1];
  stage_linear(kcur, kb, tid);          // k slab is contiguous
  stage_rows(vcur, vb, SEQ, tid);       // vT rows strided by SEQ
#pragma clang loop unroll(disable)
  for (int t0 = 0; t0 < SEQ; t0 += 32) {
    wait_async0();
    __syncthreads();
    if (t0 + 32 < SEQ) {
      stage_linear(knxt, kb + (size_t)(t0 + 32) * DKH, tid);
      stage_rows(vnxt, vb + (t0 + 32), SEQ, tid);
    }

    // scores 16x32: 4 K-tiles from LDS, then 4 WMMAs (K = DK = 64 as 2x32).
    v16h bk[4];
    bk[0] = load_b_nmajor_f16(kcur, 64, 0,  0,  lane);
    bk[1] = load_b_nmajor_f16(kcur, 64, 32, 0,  lane);
    bk[2] = load_b_nmajor_f16(kcur, 64, 0,  16, lane);
    bk[3] = load_b_nmajor_f16(kcur, 64, 32, 16, lane);
    v8f sc[2];
    {
      v8f z0 = {}, z1 = {};
      z0 = WMMA_F16(aq0, bk[0], z0);
      z0 = WMMA_F16(aq1, bk[1], z0);
      z1 = WMMA_F16(aq0, bk[2], z1);
      z1 = WMMA_F16(aq1, bk[3], z1);
#pragma unroll
      for (int r = 0; r < 8; ++r) { z0[r] *= 0.125f; z1[r] *= 0.125f; }
      sc[0] = z0; sc[1] = z1;
    }
    // Row reductions: row r in lanes 0-15 (r+8 in 16-31); xor masks 1,2,4,8
    // stay inside each 16-lane half -> per-row butterfly reduce.
    float al[8];
#pragma unroll
    for (int r = 0; r < 8; ++r) {
      float v = fmaxf(sc[0][r], sc[1][r]);
#pragma unroll
      for (int d = 1; d < 16; d <<= 1) v = fmaxf(v, __shfl_xor(v, d, 32));
      float nm = fmaxf(mrow[r], v);
      al[r] = __expf(mrow[r] - nm);
      mrow[r] = nm;
    }
#pragma unroll
    for (int r = 0; r < 8; ++r) {
      sc[0][r] = __expf(sc[0][r] - mrow[r]);
      sc[1][r] = __expf(sc[1][r] - mrow[r]);
      float v = sc[0][r] + sc[1][r];
#pragma unroll
      for (int d = 1; d < 16; d <<= 1) v += __shfl_xor(v, d, 32);
      lrow[r] = lrow[r] * al[r] + v;
    }
#pragma unroll
    for (int t = 0; t < 4; ++t)
#pragma unroll
      for (int r = 0; r < 8; ++r) o[t][r] *= al[r];

    // P: C layout -> row-major LDS -> A layout registers
#pragma unroll
    for (int r = 0; r < 8; ++r) {
      pw[(moff + r) * 32 + n]      = (h16)sc[0][r];
      pw[(moff + r) * 32 + 16 + n] = (h16)sc[1][r];
    }
    __syncthreads();
    v16h ap = load_a_f16(pw, 32, 0, 0, lane);

    // O += P (16x32) x V (32x64): 4 V-tiles from LDS, then 4 WMMAs.
    v16h bv[4];
#pragma unroll
    for (int t = 0; t < 4; ++t)
      bv[t] = load_b_nmajor_f16(vcur, 32, 0, t * 16, lane);
#pragma unroll
    for (int t = 0; t < 4; ++t)
      o[t] = WMMA_F16(ap, bv[t], o[t]);
    __syncthreads();  // reads of kbuf/vbuf/pbuf done before restage/overwrite
    h16* t1 = kcur; kcur = knxt; knxt = t1;
    h16* t2 = vcur; vcur = vnxt; vnxt = t2;
  }
  // normalize and write concatenated heads: cat[s][h*DV + dv] (f16)
#pragma unroll
  for (int t = 0; t < 4; ++t)
#pragma unroll
    for (int r = 0; r < 8; ++r) {
      float v = o[t][r] / lrow[r];
      cat[(size_t)(s0 + moff + r) * (NH * DVH) + h * DVH + t * 16 + n] = (h16)v;
    }
}

// ---------------- Kernel 3: output projection + bias ----------------
// block = 4 waves, grid = (S/64, D/64). WcT is N-major: [d][hdv].
__global__ void outproj_kernel(const h16* __restrict__ cat,
                               const h16* __restrict__ WcT,
                               const float* __restrict__ bc,
                               float* __restrict__ out) {
  __shared__ h16 wbuf[2][64 * 32];  // 8 KB ping-pong
  const int tid  = threadIdx.x;
  const int lane = tid & 31;
  const int wave = tid >> 5;
  const int s0 = blockIdx.x * 64 + wave * 16;
  const int d0 = blockIdx.y * 64;
  const h16* wg = WcT + (size_t)d0 * (NH * DVH);  // rows d0..d0+63

  v8f c[4] = {};
  h16* cur = wbuf[0];
  h16* nxt = wbuf[1];
  stage_rows(cur, wg, NH * DVH, tid);
#pragma clang loop unroll(disable)
  for (int kk = 0; kk < NH * DVH; kk += 32) {
    wait_async0();
    __syncthreads();
    if (kk + 32 < NH * DVH) stage_rows(nxt, wg + kk + 32, NH * DVH, tid);

    v16h a = load_a_f16(cat, NH * DVH, s0, kk, lane);
    v16h b[4];
#pragma unroll
    for (int t = 0; t < 4; ++t)
      b[t] = load_b_nmajor_f16(cur, 32, 0, t * 16, lane);
#pragma unroll
    for (int t = 0; t < 4; ++t)
      c[t] = WMMA_F16(a, b[t], c[t]);
    __syncthreads();
    h16* tmp = cur; cur = nxt; nxt = tmp;
  }
  const int n = lane & 15;
  const int moff = (lane < 16) ? 0 : 8;
#pragma unroll
  for (int t = 0; t < 4; ++t) {
    float bias = bc[d0 + t * 16 + n];
#pragma unroll
    for (int r = 0; r < 8; ++r)
      out[(size_t)(s0 + moff + r) * DIM + d0 + t * 16 + n] = c[t][r] + bias;
  }
}

extern "C" void kernel_launch(void* const* d_in, const int* in_sizes, int n_in,
                              void* d_out, int out_size, void* d_ws, size_t ws_size,
                              hipStream_t stream) {
  const float* x  = (const float*)d_in[0];
  const float* Wq = (const float*)d_in[1];
  const float* Wk = (const float*)d_in[2];
  const float* Wv = (const float*)d_in[3];
  const float* Wc = (const float*)d_in[4];
  const float* bc = (const float*)d_in[5];
  float* out = (float*)d_out;

  h16* qh  = (h16*)d_ws;                          // H*S*DK      (8 MB)
  h16* kh  = qh  + (size_t)NH * SEQ * DKH;        // H*S*DK      (8 MB)
  h16* vT  = kh  + (size_t)NH * SEQ * DKH;        // H*DV*S      (8 MB)
  h16* cat = vT  + (size_t)NH * DVH * SEQ;        // S*H*DV      (8 MB)
  h16* x16 = cat + (size_t)SEQ * NH * DVH;        // S*D         (8 MB)
  h16* WqT = x16 + (size_t)SEQ * DIM;             // H*DK*D      (2 MB)
  h16* WkT = WqT + (size_t)NH * DKH * DIM;        // H*DK*D      (2 MB)
  h16* WvT = WkT + (size_t)NH * DKH * DIM;        // H*DV*D      (2 MB)
  h16* WcT = WvT + (size_t)NH * DVH * DIM;        // D*(H*DV)    (2 MB)  48 MB

  // Prep: convert x; transpose+convert weights into N-major f16.
  cvt_x_kernel<<<(SEQ * DIM) / 256, 256, 0, stream>>>(x, x16);
  tr_cvt_kernel<<<dim3(DKH / 32, DIM / 32, NH), 256, 0, stream>>>(
      Wq, WqT, DIM, DKH, (size_t)DIM * DKH, (size_t)DKH * DIM);
  tr_cvt_kernel<<<dim3(DKH / 32, DIM / 32, NH), 256, 0, stream>>>(
      Wk, WkT, DIM, DKH, (size_t)DIM * DKH, (size_t)DKH * DIM);
  tr_cvt_kernel<<<dim3(DVH / 32, DIM / 32, NH), 256, 0, stream>>>(
      Wv, WvT, DIM, DVH, (size_t)DIM * DVH, (size_t)DVH * DIM);
  tr_cvt_kernel<<<dim3(DIM / 32, (NH * DVH) / 32, 1), 256, 0, stream>>>(
      Wc, WcT, NH * DVH, DIM, 0, 0);

  // Main pipeline.
  qkv_proj_kernel<<<dim3(SEQ / 64, NH, 3), 128, 0, stream>>>(x16, WqT, WkT, WvT,
                                                             qh, kh, vT);
  attn_kernel<<<dim3(SEQ / 64, NH), 128, 0, stream>>>(qh, kh, vT, cat);
  outproj_kernel<<<dim3(SEQ / 64, DIM / 64), 128, 0, stream>>>(cat, WcT, bc, out);
}
